// GraphAttentionNetwork_12275016532267
// MI455X (gfx1250) — matrix operand order repaired and checked by reading, not compiled, not run on hardware
//
#include <hip/hip_runtime.h>
#include <hip/hip_bf16.h>

typedef __attribute__((ext_vector_type(16))) _Float16 v16h;
typedef __attribute__((ext_vector_type(8)))  _Float16 v8h;
typedef __attribute__((ext_vector_type(8)))  float    v8f;

#define NEG_SLOPE 0.2f
#define BN_EPS    1e-5f

static inline int cdiv(int a, int b) { return (a + b - 1) / b; }

// ---------------------------------------------------------------------------
// helpers
// ---------------------------------------------------------------------------
__device__ __forceinline__ void edge_sd(const int* __restrict__ ei, int E, int e,
                                        int& s, int& d) {
  if (e < E) { s = ei[e]; d = ei[E + e]; }     // row0 = src, row1 = dst
  else       { s = e - E; d = e - E; }          // implicit self loops
}

__device__ __forceinline__ void atomic_max_f32(float* addr, float val) {
  // standard monotone bit-pattern trick; amax initialized to -FLT_MAX
  if (val >= 0.0f) atomicMax((int*)addr, __float_as_int(val));
  else             atomicMin((unsigned int*)addr, __float_as_uint(val));
}

__device__ __forceinline__ float lrelu(float x) {
  return x >= 0.0f ? x : NEG_SLOPE * x;
}

// ---------------------------------------------------------------------------
// generic WMMA f16 GEMM:  C[M,N] = A[M,K] * Bt[N,K]^T (+bias[N])
// requires M%16==0, N%64==0, K%32==0.  One wave -> 16x64 output strip.
// ---------------------------------------------------------------------------
__global__ void __launch_bounds__(256)
gemm_wmma_f16(const _Float16* __restrict__ A, const _Float16* __restrict__ Bt,
              const float* __restrict__ bias, float* __restrict__ C,
              int M, int N, int K)
{
  const int wave   = (blockIdx.x * blockDim.x + threadIdx.x) >> 5;
  const int lane   = threadIdx.x & 31;
  const int mTiles = M >> 4;
  const int nStrips = N >> 6;
  if (wave >= mTiles * nStrips) return;          // wave-uniform: EXEC stays all-1s
  const int mT = wave % mTiles;
  const int nS = wave / mTiles;
  const int r  = lane & 15;
  const int hi = lane >> 4;

  const _Float16* arow  = A  + (size_t)(mT * 16 + r) * K;        // A row, row-major
  const _Float16* bbase = Bt + (size_t)(nS * 64 + r) * K + hi * 16;

  v8f acc[4] = {};
  for (int kb = 0; kb < K; kb += 32) {
    // A fragment (16x32 f16): lane r+16*hi -> row r, K = kb + 16*hi + j
    union { v16h v; v8h h[2]; } a;
    a.v = *(const v16h*)(arow + kb + 16 * hi);
#pragma unroll
    for (int t = 0; t < 4; ++t) {
      // B fragment (32x16 f16): lane -> col (lane&15), K = kb + 16*hk + j
      v16h b = *(const v16h*)(bbase + (size_t)t * 16 * K + kb);
      acc[t] = __builtin_amdgcn_wmma_f32_16x16x32_f16(
          false, a.v, false, b, (short)0, acc[t], false, false);
    }
  }
#pragma unroll
  for (int t = 0; t < 4; ++t) {
    const int col = nS * 64 + t * 16 + r;
    const float bs = bias ? bias[col] : 0.0f;
    float* crow = C + (size_t)(mT * 16 + 8 * hi) * N + col;   // VGPR i -> row i+8*hi
#pragma unroll
    for (int i = 0; i < 8; ++i) crow[(size_t)i * N] = acc[t][i] + bs;
  }
}

// ---------------------------------------------------------------------------
// conversions / fills
// ---------------------------------------------------------------------------
__global__ void f32_to_f16(const float* __restrict__ s, _Float16* __restrict__ d, int n) {
  int t = blockIdx.x * blockDim.x + threadIdx.x;
  if (t < n) d[t] = (_Float16)s[t];
}

__global__ void w_transpose_f16(const float* __restrict__ W, _Float16* __restrict__ Wt,
                                int K, int N) {   // W: KxN f32 -> Wt: NxK f16
  int t = blockIdx.x * blockDim.x + threadIdx.x;
  if (t >= K * N) return;
  int k = t / N, n = t % N;
  Wt[(size_t)n * K + k] = (_Float16)W[t];
}

__global__ void fill_f32(float* __restrict__ p, float v, int n) {
  int t = blockIdx.x * blockDim.x + threadIdx.x;
  if (t < n) p[t] = v;
}

// ---------------------------------------------------------------------------
// attention logits: es[n,h] = <hW[n,h,:], a_src[h,:]>, ed likewise. wave per (n,h)
// ---------------------------------------------------------------------------
__global__ void __launch_bounds__(256)
gat_dots(const float* __restrict__ hW, const float* __restrict__ a_s,
         const float* __restrict__ a_d, float* __restrict__ es,
         float* __restrict__ ed, int Nn, int H, int C)
{
  int wave = (blockIdx.x * blockDim.x + threadIdx.x) >> 5;
  int lane = threadIdx.x & 31;
  if (wave >= Nn * H) return;
  int n = wave / H, h = wave % H;
  const float* hp = hW + (size_t)n * H * C + (size_t)h * C;
  const float* ap = a_s + (size_t)h * C;
  const float* dp = a_d + (size_t)h * C;
  float s = 0.0f, d = 0.0f;
  for (int c = lane; c < C; c += 32) { float v = hp[c]; s += v * ap[c]; d += v * dp[c]; }
  for (int off = 16; off; off >>= 1) { s += __shfl_xor(s, off); d += __shfl_xor(d, off); }
  if (lane == 0) { es[wave] = s; ed[wave] = d; }
}

// ---------------------------------------------------------------------------
// edge passes
// ---------------------------------------------------------------------------
__global__ void edge_max(const int* __restrict__ ei, int E, int Etot, int H,
                         const float* __restrict__ es, const float* __restrict__ ed,
                         float* __restrict__ amax)
{
  int t = blockIdx.x * blockDim.x + threadIdx.x;
  if (t >= Etot * H) return;
  int e = t / H, h = t % H, s, d;
  edge_sd(ei, E, e, s, d);
  float a = lrelu(es[s * H + h] + ed[d * H + h]);
  atomic_max_f32(&amax[d * H + h], a);
}

__global__ void edge_exp(const int* __restrict__ ei, int E, int Etot, int H,
                         const float* __restrict__ es, const float* __restrict__ ed,
                         const float* __restrict__ amax, float* __restrict__ exb,
                         float* __restrict__ den)
{
  int t = blockIdx.x * blockDim.x + threadIdx.x;
  if (t >= Etot * H) return;
  int e = t / H, h = t % H, s, d;
  edge_sd(ei, E, e, s, d);
  float a  = lrelu(es[s * H + h] + ed[d * H + h]);
  float ex = expf(a - amax[d * H + h]);
  exb[t] = ex;
  atomicAdd(&den[d * H + h], ex);
}

// agg[dst,h,c] += hW[src,h,c] * w[e,h] ; thread per (edge, 4-channel chunk)
__global__ void __launch_bounds__(256)
edge_agg(const int* __restrict__ ei, int E, int Etot, int H, int C,
         const float* __restrict__ hW, const float* __restrict__ exb,
         const float* __restrict__ den, float* __restrict__ agg)
{
  const int HC = H * C;
  const int chunks = HC >> 2;            // float4 chunks per edge (<=256)
  const int epb = 256 / chunks;          // edges per block
  const int le = threadIdx.x / chunks;
  const int ch = threadIdx.x % chunks;
  const int e  = blockIdx.x * epb + le;
  if (e >= Etot) return;
  int s, d; edge_sd(ei, E, e, s, d);
  const int h = (ch << 2) / C;
  const float w = exb[e * H + h] / (den[d * H + h] + 1e-16f);
  const float4 v = *(const float4*)(hW + (size_t)s * HC + (ch << 2));
  float* o = agg + (size_t)d * HC + (ch << 2);
  atomicAdd(o + 0, v.x * w);
  atomicAdd(o + 1, v.y * w);
  atomicAdd(o + 2, v.z * w);
  atomicAdd(o + 3, v.w * w);
}

// ---------------------------------------------------------------------------
// node epilogues
// ---------------------------------------------------------------------------
__global__ void head_mean(const float* __restrict__ agg, const float* __restrict__ b,
                          float* __restrict__ out, int Nn, int H, int C)
{
  int t = blockIdx.x * blockDim.x + threadIdx.x;
  if (t >= Nn * C) return;
  int n = t / C, c = t % C;
  float s = 0.0f;
  for (int h = 0; h < H; ++h) s += agg[(size_t)n * H * C + (size_t)h * C + c];
  out[t] = s * (1.0f / H) + b[c];
}

__global__ void bn_stats(const float* __restrict__ x, float* __restrict__ sum,
                         float* __restrict__ sumsq, int Nn, int C)
{
  int c = threadIdx.x;            // blockDim.x == C
  float s = 0.0f, q = 0.0f;
  for (int n = blockIdx.x; n < Nn; n += gridDim.x) {
    float v = x[(size_t)n * C + c];
    s += v; q += v * v;
  }
  atomicAdd(&sum[c], s);
  atomicAdd(&sumsq[c], q);
}

__global__ void bn_final(const float* __restrict__ sum, const float* __restrict__ sumsq,
                         const float* __restrict__ g, const float* __restrict__ be,
                         float* __restrict__ scl, float* __restrict__ sft, int Nn, int C)
{
  int c = threadIdx.x;
  if (c >= C) return;
  float inv = 1.0f / (float)Nn;
  float mu  = sum[c] * inv;
  float var = sumsq[c] * inv - mu * mu;
  float sc  = g[c] * rsqrtf(var + BN_EPS);
  scl[c] = sc;
  sft[c] = be[c] - mu * sc;
}

__global__ void bn_elu_res(float* __restrict__ io, const float* __restrict__ scl,
                           const float* __restrict__ sft, const float* __restrict__ res,
                           int Nn, int C)
{
  int t = blockIdx.x * blockDim.x + threadIdx.x;
  if (t >= Nn * C) return;
  int c = t % C;
  float v = io[t] * scl[c] + sft[c];
  v = v > 0.0f ? v : expm1f(v);
  io[t] = v + res[t];
}

// ---------------------------------------------------------------------------
// host-side orchestration
// ---------------------------------------------------------------------------
static void launch_gemm(const _Float16* A, const _Float16* Bt, const float* bias,
                        float* C, int M, int N, int K, hipStream_t stream)
{
  int waves = (M / 16) * (N / 64);
  gemm_wmma_f16<<<cdiv(waves * 32, 256), 256, 0, stream>>>(A, Bt, bias, C, M, N, K);
}

static void gat_layer(const float* h, const float* W, const float* a_s, const float* a_d,
                      const float* b, const float* g, const float* be,
                      const float* res, float* out,
                      int Nn, const int* ei, int E, int Etot, int H, int C,
                      _Float16* A16, _Float16* Wt, float* hW, float* agg,
                      float* es, float* ed, float* amax, float* den, float* exb,
                      float* csum, float* csq, float* scl, float* sft,
                      hipStream_t stream)
{
  const int K = 128, HC = H * C;
  w_transpose_f16<<<cdiv(K * HC, 256), 256, 0, stream>>>(W, Wt, K, HC);
  f32_to_f16<<<cdiv(Nn * K, 256), 256, 0, stream>>>(h, A16, Nn * K);
  launch_gemm(A16, Wt, nullptr, hW, Nn, HC, K, stream);

  gat_dots<<<cdiv(Nn * H * 32, 256), 256, 0, stream>>>(hW, a_s, a_d, es, ed, Nn, H, C);

  fill_f32<<<cdiv(Nn * H, 256), 256, 0, stream>>>(amax, -3.402823e38f, Nn * H);
  fill_f32<<<cdiv(Nn * H, 256), 256, 0, stream>>>(den, 0.0f, Nn * H);
  fill_f32<<<cdiv(Nn * HC, 256), 256, 0, stream>>>(agg, 0.0f, Nn * HC);

  edge_max<<<cdiv(Etot * H, 256), 256, 0, stream>>>(ei, E, Etot, H, es, ed, amax);
  edge_exp<<<cdiv(Etot * H, 256), 256, 0, stream>>>(ei, E, Etot, H, es, ed, amax, exb, den);
  const int epb = 256 / (HC >> 2);
  edge_agg<<<cdiv(Etot, epb), 256, 0, stream>>>(ei, E, Etot, H, C, hW, exb, den, agg);

  head_mean<<<cdiv(Nn * C, 256), 256, 0, stream>>>(agg, b, out, Nn, H, C);

  fill_f32<<<1, 256, 0, stream>>>(csum, 0.0f, C);
  fill_f32<<<1, 256, 0, stream>>>(csq, 0.0f, C);
  bn_stats<<<256, C, 0, stream>>>(out, csum, csq, Nn, C);
  bn_final<<<1, C, 0, stream>>>(csum, csq, g, be, scl, sft, Nn, C);
  bn_elu_res<<<cdiv(Nn * C, 256), 256, 0, stream>>>(out, scl, sft, res, Nn, C);
}

extern "C" void kernel_launch(void* const* d_in, const int* in_sizes, int n_in,
                              void* d_out, int out_size, void* d_ws, size_t ws_size,
                              hipStream_t stream)
{
  const float* x   = (const float*)d_in[0];
  const int*   ei  = (const int*)d_in[1];
  const float* Wp  = (const float*)d_in[2];
  const float* bp  = (const float*)d_in[3];
  const float* W0  = (const float*)d_in[4];
  const float* as0 = (const float*)d_in[5];
  const float* ad0 = (const float*)d_in[6];
  const float* b0  = (const float*)d_in[7];
  const float* g0  = (const float*)d_in[8];
  const float* be0 = (const float*)d_in[9];
  const float* W1  = (const float*)d_in[10];
  const float* as1 = (const float*)d_in[11];
  const float* ad1 = (const float*)d_in[12];
  const float* b1  = (const float*)d_in[13];
  const float* g1  = (const float*)d_in[14];
  const float* be1 = (const float*)d_in[15];
  const float* W2  = (const float*)d_in[16];
  const float* as2 = (const float*)d_in[17];
  const float* ad2 = (const float*)d_in[18];
  const float* b2  = (const float*)d_in[19];
  const float* g2  = (const float*)d_in[20];
  const float* be2 = (const float*)d_in[21];
  const float* Wr  = (const float*)d_in[22];
  const float* br  = (const float*)d_in[23];
  float* out = (float*)d_out;
  (void)n_in; (void)out_size; (void)ws_size;

  const int Nn   = in_sizes[0] / 64;
  const int E    = in_sizes[1] / 2;
  const int Etot = E + Nn;

  char* p = (char*)d_ws;
  auto carve = [&](size_t bytes) -> char* {
    char* q = p; p += (bytes + 255) & ~(size_t)255; return q;
  };
  float*    hA   = (float*)carve((size_t)Nn * 128 * sizeof(float));
  float*    hB   = (float*)carve((size_t)Nn * 128 * sizeof(float));
  float*    hr   = (float*)carve((size_t)Nn * 64 * sizeof(float));
  float*    hW   = (float*)carve((size_t)Nn * 1024 * sizeof(float));
  float*    agg  = (float*)carve((size_t)Nn * 1024 * sizeof(float));
  _Float16* A16  = (_Float16*)carve((size_t)Nn * 128 * sizeof(_Float16));
  _Float16* Wt   = (_Float16*)carve((size_t)128 * 1024 * sizeof(_Float16));
  float*    es   = (float*)carve((size_t)Nn * 8 * sizeof(float));
  float*    edv  = (float*)carve((size_t)Nn * 8 * sizeof(float));
  float*    amax = (float*)carve((size_t)Nn * 8 * sizeof(float));
  float*    den  = (float*)carve((size_t)Nn * 8 * sizeof(float));
  float*    exb  = (float*)carve((size_t)Etot * 8 * sizeof(float));
  float*    csum = (float*)carve(128 * sizeof(float));
  float*    csq  = (float*)carve(128 * sizeof(float));
  float*    scl  = (float*)carve(128 * sizeof(float));
  float*    sft  = (float*)carve(128 * sizeof(float));

  // Stage 0: h = x @ Wp + bp   (Nn x 64) @ (64 x 128)
  w_transpose_f16<<<cdiv(64 * 128, 256), 256, 0, stream>>>(Wp, Wt, 64, 128);
  f32_to_f16<<<cdiv(Nn * 64, 256), 256, 0, stream>>>(x, A16, Nn * 64);
  launch_gemm(A16, Wt, bp, hA, Nn, 128, 64, stream);

  // Layer 0 (8 heads, identity residual): hB = elu(bn(gat(hA))) + hA
  gat_layer(hA, W0, as0, ad0, b0, g0, be0, hA, hB, Nn, ei, E, Etot, 8, 128,
            A16, Wt, hW, agg, es, edv, amax, den, exb, csum, csq, scl, sft, stream);
  // Layer 1: hA = elu(bn(gat(hB))) + hB
  gat_layer(hB, W1, as1, ad1, b1, g1, be1, hB, hA, Nn, ei, E, Etot, 8, 128,
            A16, Wt, hW, agg, es, edv, amax, den, exb, csum, csq, scl, sft, stream);

  // Layer 2 residual projection: hr = hA @ Wr + br   (128 -> 64)
  w_transpose_f16<<<cdiv(128 * 64, 256), 256, 0, stream>>>(Wr, Wt, 128, 64);
  f32_to_f16<<<cdiv(Nn * 128, 256), 256, 0, stream>>>(hA, A16, Nn * 128);
  launch_gemm(A16, Wt, br, hr, Nn, 64, 128, stream);

  // Layer 2 (1 head, out=64): out = elu(bn(gat(hA))) + hr
  gat_layer(hA, W2, as2, ad2, b2, g2, be2, hr, out, Nn, ei, E, Etot, 1, 64,
            A16, Wt, hW, agg, es, edv, amax, den, exb, csum, csq, scl, sft, stream);
}
